// Net_LV_Improved_31061203484782
// MI455X (gfx1250) — matrix operand order repaired and checked by reading, not compiled, not run on hardware
//
#include <hip/hip_runtime.h>
#include <math.h>

// ---------------------------------------------------------------------------
// CDNA5 (gfx1250) implementation of the Net_LV MC forward pass.
// wave32, WMMA f32_16x16x32_f16 for all MLP layers, TDM weight staging,
// bias folded into the GEMM via an augmented bias-lane (activation col == 1),
// packed f16 epilogue (v_cvt_pk_rtz) + raw v_max_num relu via inline asm.
// ---------------------------------------------------------------------------

typedef __attribute__((ext_vector_type(16))) _Float16 v16h;
typedef __attribute__((ext_vector_type(2)))  _Float16 v2h;
typedef __attribute__((ext_vector_type(8)))  float    v8f;
typedef unsigned int v4u __attribute__((ext_vector_type(4)));
typedef int          v8i __attribute__((ext_vector_type(8)));
typedef int          v4i __attribute__((ext_vector_type(4)));

#define MC        16384
#define NSTEPS    120
#define PERIOD    30
#define NMAT      4
#define NSTR      21
#define RRATE     0.05f

// Padded dims:
//  d-net: in 2->32 (K, bias lane col 2), width 50->64 (bias lane col 50), out 1->16
//  v-net: in 2->32, width 30->32 (bias lane col 30), out 84->96
//  e-net: in 122->128 (bias lane col 122), width 20->32 (bias lane col 20), out 1->16
#define OFF_DW0   0        // 32*64
#define OFF_DB0   2048     // 64   (unused)
#define OFF_DH    2112     // 3*64*64
#define OFF_DHB   14400    // 3*64 (unused)
#define OFF_DWO   14592    // 64*16
#define OFF_DBO   15616    // 16   (unused)
#define OFF_VW0   15632    // 32*32
#define OFF_VB0   16656    // 32   (unused)
#define OFF_VH    16688    // 3*32*32
#define OFF_VHB   19760    // 3*32 (unused)
#define OFF_VWO   19856    // 32*96
#define OFF_VBO   22928    // 96   (unused)
#define OFF_EW0   23024    // 128*32
#define OFF_EB0   27120    // 32   (unused)
#define OFF_EH    27152    // 3*32*32
#define OFF_EHB   30224    // 3*32 (unused)
#define OFF_EWO   30320    // 32*16
#define OFF_EBO   30832    // 16   (unused)
#define SEG_HALVES 30848   // halves per segment (61696 bytes)
#define SEG_QW     (SEG_HALVES / 4)  // 7712 8-byte elements per segment

// Workspace layout (bytes)
#define WS_W      0                         // 4*SEG_HALVES*2 = 246784
#define WS_ACC    246784                    // 171 floats: pr[84], pr2[84], E, PE, PE2
#define WS_EVAL   247808                    // MC floats
#define WS_CVE    313344                    // MC floats

// LDS layout (bytes) — dynamic shared memory, 320KB/WGP available on CDNA5
#define LDS_WSEG   0            // 61696
#define LDS_ACTIN  61696        // 64*32*2 = 4096
#define LDS_BUFA   65792        // 64*64*2 = 8192
#define LDS_BUFB   73984        // 8192
#define LDS_EINP   82176        // 64*128*2 = 16384
#define LDS_DOUTS  98560        // 64 f32
#define LDS_EOUTS  98816
#define LDS_FAC    99072
#define LDS_SNEW   99328
#define LDS_TOTAL  99584

#if __has_builtin(__builtin_amdgcn_tensor_load_to_lds) && \
    __has_builtin(__builtin_amdgcn_s_wait_tensorcnt)
#define USE_TDM 1
#else
#define USE_TDM 0
#endif

// relu as exactly one v_max_num_f32 (inline constant 0 in SRC0, VOP2-legal);
// avoids the canonicalize the compiler inserts for IEEE fmax lowering.
__device__ __forceinline__ float relu1(float x) {
  float r;
  asm("v_max_num_f32 %0, 0, %1" : "=v"(r) : "v"(x));
  return r;
}

// pack two f32 into packed f16 (v_cvt_pk_rtz_f16_f32), bit-cast to _Float16x2
__device__ __forceinline__ v2h pack_h2(float x0, float x1) {
  return __builtin_bit_cast(v2h, __builtin_amdgcn_cvt_pkrtz(x0, x1));
}

// --------------------------- WMMA fragment loads ---------------------------
// A-matrix 16x32 f16 (ISA 7.12.2): lane l holds row m=l&15;
// halves 0..7 = K[k0 + (l>>4)*8 .. +7], halves 8..15 = K[k0+16+(l>>4)*8 ..]
__device__ __forceinline__ v16h load_a_frag(const _Float16* src, int lane,
                                            int kbase, int stride) {
  const int m  = lane & 15;
  const int hh = lane >> 4;
  const _Float16* row = src + m * stride + kbase + hh * 8;
  v16h a;
#pragma unroll
  for (int j = 0; j < 8; ++j) a[j] = row[j];
#pragma unroll
  for (int j = 0; j < 8; ++j) a[8 + j] = row[16 + j];
  return a;
}

// B-matrix 32x16 f16: lane l holds col n = ntile*16 + (l&15);
// halves 0..7 = K[k0 + (l>>4)*8 .. +7], halves 8..15 = K[k0+16+(l>>4)*8 ..]
__device__ __forceinline__ v16h load_b_frag(const _Float16* W, int lane,
                                            int kbase, int ntile, int Np) {
  const int hh = lane >> 4;
  const _Float16* p = W + (kbase + hh * 8) * Np + ntile * 16 + (lane & 15);
  v16h b;
#pragma unroll
  for (int j = 0; j < 8; ++j) b[j] = p[j * Np];
#pragma unroll
  for (int j = 0; j < 8; ++j) b[8 + j] = p[(16 + j) * Np];
  return b;
}

// D = relu?(A @ W); bias folded into W (augmented bias lane in activations).
// src row-major [16][srcStride] f16 (wave-local), dst row-major
// [16][dstStride] f16. D-layout: lane l -> col n=l&15, rows m = 8*(l>>4)+r.
// Epilogue: relu in f32 (1 op), then v_cvt_pk_rtz packs 2 rows per VALU op.
template <int KT, int NT, bool RELU>
__device__ __forceinline__ void mlp_layer(const _Float16* src, int srcStride,
                                          const _Float16* W, int Np,
                                          _Float16* dst, int dstStride,
                                          int lane) {
#pragma unroll
  for (int nt = 0; nt < NT; ++nt) {
    v8f acc;
#pragma unroll
    for (int r = 0; r < 8; ++r) acc[r] = 0.f;
#pragma unroll
    for (int kt = 0; kt < KT; ++kt) {
      v16h a = load_a_frag(src, lane, kt * 32, srcStride);
      v16h b = load_b_frag(W, lane, kt * 32, nt, Np);
      acc = __builtin_amdgcn_wmma_f32_16x16x32_f16(false, a, false, b,
                                                   (short)0, acc, false, false);
    }
    const int n     = nt * 16 + (lane & 15);
    const int mbase = (lane >> 4) << 3;
#pragma unroll
    for (int rp = 0; rp < 4; ++rp) {
      float x0 = acc[2 * rp], x1 = acc[2 * rp + 1];
      if (RELU) { x0 = relu1(x0); x1 = relu1(x1); }
      v2h p = pack_h2(x0, x1);
      dst[(mbase + 2 * rp) * dstStride + n]     = p[0];
      dst[(mbase + 2 * rp + 1) * dstStride + n] = p[1];
    }
  }
}

// Output layer, keep only column 0 (true output dim 1), write f32 per sample.
template <int KT>
__device__ __forceinline__ void mlp_out_col0(const _Float16* src, int srcStride,
                                             const _Float16* W,
                                             float* dcol, int lane) {
  v8f acc;
#pragma unroll
  for (int r = 0; r < 8; ++r) acc[r] = 0.f;
#pragma unroll
  for (int kt = 0; kt < KT; ++kt) {
    v16h a = load_a_frag(src, lane, kt * 32, srcStride);
    v16h b = load_b_frag(W, lane, kt * 32, 0, 16);
    acc = __builtin_amdgcn_wmma_f32_16x16x32_f16(false, a, false, b,
                                                 (short)0, acc, false, false);
  }
  if ((lane & 15) == 0) {
    const int mbase = (lane >> 4) << 3;
#pragma unroll
    for (int r = 0; r < 8; ++r) dcol[mbase + r] = acc[r];
  }
}

// --------------------------- weight prep kernel ----------------------------
// Pads W [K][N] -> [Kp][Np] f16 with bias at row K, and (optionally) a 1.0 at
// (K, N) so the augmented bias-lane propagates through relu to the next layer.
__device__ __forceinline__ void pad2b(_Float16* dst, const float* W,
                                      const float* b, int K, int N, int Kp,
                                      int Np, bool prop, int tid, int nth) {
  for (int x = tid; x < Kp * Np; x += nth) {
    const int k = x / Np, n = x % Np;
    float v = 0.f;
    if (k < K && n < N) v = W[k * N + n];
    else if (k == K && n < N) v = b[n];
    else if (prop && k == K && n == N) v = 1.f;
    dst[x] = (_Float16)v;
  }
}

__global__ void prep_weights(
    const float* dW0, const float* db0, const float* dhW, const float* dhb,
    const float* dWo, const float* dbo, const float* vW0, const float* vb0,
    const float* vhW, const float* vhb, const float* vWo, const float* vbo,
    const float* eW0, const float* eb0, const float* ehW, const float* ehb,
    const float* eWo, const float* ebo, _Float16* wsW) {
  const int mat = blockIdx.x;
  const int tid = threadIdx.x, nth = blockDim.x;
  _Float16* seg = wsW + (size_t)mat * SEG_HALVES;
  // d-net
  pad2b(seg + OFF_DW0, dW0 + mat * 2 * 50, db0 + mat * 50, 2, 50, 32, 64, true, tid, nth);
  for (int L = 0; L < 3; ++L)
    pad2b(seg + OFF_DH + L * 4096, dhW + (mat * 3 + L) * 2500,
          dhb + (mat * 3 + L) * 50, 50, 50, 64, 64, true, tid, nth);
  pad2b(seg + OFF_DWO, dWo + mat * 50, dbo + mat, 50, 1, 64, 16, false, tid, nth);
  // v-net
  pad2b(seg + OFF_VW0, vW0 + mat * 2 * 30, vb0 + mat * 30, 2, 30, 32, 32, true, tid, nth);
  for (int L = 0; L < 3; ++L)
    pad2b(seg + OFF_VH + L * 1024, vhW + (mat * 3 + L) * 900,
          vhb + (mat * 3 + L) * 30, 30, 30, 32, 32, true, tid, nth);
  pad2b(seg + OFF_VWO, vWo + mat * 30 * 84, vbo + mat * 84, 30, 84, 32, 96, false, tid, nth);
  // e-net
  pad2b(seg + OFF_EW0, eW0 + mat * 122 * 20, eb0 + mat * 20, 122, 20, 128, 32, true, tid, nth);
  for (int L = 0; L < 3; ++L)
    pad2b(seg + OFF_EH + L * 1024, ehW + (mat * 3 + L) * 400,
          ehb + (mat * 3 + L) * 20, 20, 20, 32, 32, true, tid, nth);
  pad2b(seg + OFF_EWO, eWo + mat * 20, ebo + mat, 20, 1, 32, 16, false, tid, nth);
}

__global__ void init_acc(float* acc) {
  const int t = blockIdx.x * blockDim.x + threadIdx.x;
  if (t < 171) acc[t] = 0.f;
}

// ------------------------------ main MC kernel ------------------------------
__global__ __launch_bounds__(128) void mc_forward(
    const float* __restrict__ S0p, const float* __restrict__ z,
    const _Float16* __restrict__ wsW, float* __restrict__ acc,
    float* __restrict__ wsEval, float* __restrict__ wsCve) {
  extern __shared__ char smem[];
  _Float16* wseg  = (_Float16*)(smem + LDS_WSEG);
  _Float16* actIn = (_Float16*)(smem + LDS_ACTIN);  // [64][32]
  _Float16* bufA  = (_Float16*)(smem + LDS_BUFA);   // [64][64]
  _Float16* bufB  = (_Float16*)(smem + LDS_BUFB);   // [64][64]
  _Float16* einp  = (_Float16*)(smem + LDS_EINP);   // [64][128]
  float* douts = (float*)(smem + LDS_DOUTS);        // [64]
  float* eouts = (float*)(smem + LDS_EOUTS);        // [64]
  float* facv  = (float*)(smem + LDS_FAC);          // [64]
  float* snew  = (float*)(smem + LDS_SNEW);         // [64]

  const int tid  = threadIdx.x;
  const int wave = tid >> 5;
  const int lane = tid & 31;
  const int sL   = wave * 16 + (lane & 15);         // block-local sample (lanes 0-15)
  const int gs   = blockIdx.x * 64 + sL;            // global sample

  const float S0v = S0p[0];
  const float hstep = 1.0f / NSTEPS;
  const float sqh = sqrtf(hstep);

  // init per-block LDS
  for (int x = tid; x < 64 * 32; x += 128) actIn[x] = (_Float16)0.f;
  for (int x = tid; x < 64 * 128; x += 128) einp[x] = (_Float16)0.f;
  __syncthreads();
  if (lane < 16) {
    einp[sL * 128 + 1]   = (_Float16)S0v;  // path[0] = S0
    einp[sL * 128 + 122] = (_Float16)1.f;  // e-net bias lane
    actIn[sL * 32 + 2]   = (_Float16)1.f;  // d/v-net bias lane
  }

  float S_old = S0v, run_max = S0v, cv_e = 0.f, dWn = 0.f, S_new = S0v;
  float cvv[6][8];  // cv_v in WMMA D-fragments (96 padded columns)
#pragma unroll
  for (int t = 0; t < 6; ++t)
#pragma unroll
    for (int r = 0; r < 8; ++r) cvv[t][r] = 0.f;

  const _Float16* aIn = actIn + wave * 16 * 32;
  _Float16* bA = bufA + wave * 16 * 64;
  _Float16* bB = bufB + wave * 16 * 64;
  const _Float16* eIn = einp + wave * 16 * 128;

  for (int i = 1; i <= NSTEPS; ++i) {
    const int idx = (i - 1) / PERIOD;
    if (((i - 1) % PERIOD) == 0) {  // stage this segment's weights into LDS
      __syncthreads();
#if USE_TDM
      if (wave == 0) {
        // Tensor Data Mover: 1-D tile of 7712 x 8B from global to LDS 0.
        const unsigned long long ga =
            (unsigned long long)(wsW + (size_t)idx * SEG_HALVES);
        v4u g0;
        g0[0] = 1u;                                  // count=1 (valid user D#)
        g0[1] = (unsigned)LDS_WSEG;                  // lds_addr
        g0[2] = (unsigned)(ga & 0xffffffffu);        // global_addr[31:0]
        g0[3] = (unsigned)((ga >> 32) & 0x01ffffffu) // global_addr[56:32]
              | 0x80000000u;                         // type=2 (bits 127:126)
        v8i g1;
        g1[0] = (3 << 16);                           // data_size = 8 bytes
        g1[1] = (int)((SEG_QW & 0xffff) << 16);      // tensor_dim0[15:0]
        g1[2] = (int)((SEG_QW >> 16) & 0xffff) | (1 << 16); // dim0 hi | tensor_dim1=1
        g1[3] = (int)((SEG_QW & 0xffff) << 16);      // tile_dim0 (bits 127:112)
        g1[4] = 0;                                   // tile_dim1=0, tile_dim2=0
        g1[5] = (int)SEG_QW;                         // tensor_dim0_stride[31:0]
        g1[6] = 0;                                   // stride hi, dim1_stride lo
        g1[7] = 0;
        v4i gz = {0, 0, 0, 0};
#if defined(__clang_major__) && __clang_major__ >= 23
        v8i gz8 = {0, 0, 0, 0, 0, 0, 0, 0};
        __builtin_amdgcn_tensor_load_to_lds(g0, g1, gz, gz, gz8, 0);
#else
        __builtin_amdgcn_tensor_load_to_lds(g0, g1, gz, gz, 0);
#endif
        __builtin_amdgcn_s_wait_tensorcnt(0);
      }
#else
      {
        const uint4* src = (const uint4*)(wsW + (size_t)idx * SEG_HALVES);
        uint4* dst = (uint4*)wseg;
        for (int x = tid; x < SEG_HALVES * 2 / 16; x += 128) dst[x] = src[x];
      }
#endif
      __syncthreads();
    }
    const float t0 = (i - 1) * hstep;

    if (lane < 16) {  // scalar phase A: inputs for this step
      dWn = sqh * z[gs * NSTEPS + (i - 1)];
      actIn[sL * 32 + 0] = (_Float16)t0;
      actIn[sL * 32 + 1] = (_Float16)S_old;
      einp[sL * 128 + 0] = (_Float16)t0;
      if (i < NSTEPS) __builtin_prefetch(&z[gs * NSTEPS + i], 0, 0);
    }
    __syncthreads();

    // ---- d-net: (t,S,1) -> 64 -> 64 -> 64 -> 64 -> softplus scalar ----
    mlp_layer<1, 4, true>(aIn, 32, wseg + OFF_DW0, 64, bA, 64, lane);
    mlp_layer<2, 4, true>(bA, 64, wseg + OFF_DH + 0 * 4096, 64, bB, 64, lane);
    mlp_layer<2, 4, true>(bB, 64, wseg + OFF_DH + 1 * 4096, 64, bA, 64, lane);
    mlp_layer<2, 4, true>(bA, 64, wseg + OFF_DH + 2 * 4096, 64, bB, 64, lane);
    mlp_out_col0<2>(bB, 64, wseg + OFF_DWO, douts + wave * 16, lane);
    __syncthreads();

    if (lane < 16) {  // scalar phase B: SDE update + CV factor
      const float y = douts[sL];
      const float diff = (y > 0.f) ? (y + log1pf(expf(-y))) : log1pf(expf(y));
      const float Sd = S_old;
      S_new = S_old + RRATE * S_old * hstep / (1.f + RRATE * Sd * sqh)
                    + S_old * diff * dWn / (1.f + Sd * diff * sqh);
      const float disc = expf(-RRATE * t0);
      facv[sL] = disc * Sd * diff * dWn;
      snew[sL] = S_new;
      run_max = fmaxf(run_max, S_new);
    }
    __syncthreads();

    // ---- v-net: (t,S,1) -> 32 -> 32 -> 32 -> 32 -> 96 (cv_v fragments) ----
    mlp_layer<1, 2, true>(aIn, 32, wseg + OFF_VW0, 32, bA, 64, lane);
    mlp_layer<1, 2, true>(bA, 64, wseg + OFF_VH + 0 * 1024, 32, bB, 64, lane);
    mlp_layer<1, 2, true>(bB, 64, wseg + OFF_VH + 1 * 1024, 32, bA, 64, lane);
    mlp_layer<1, 2, true>(bA, 64, wseg + OFF_VH + 2 * 1024, 32, bB, 64, lane);
    {
      v16h a = load_a_frag(bB, lane, 0, 64);
#pragma unroll
      for (int nt = 0; nt < 6; ++nt) {
        v8f vo;
#pragma unroll
        for (int r = 0; r < 8; ++r) vo[r] = 0.f;
        v16h b = load_b_frag(wseg + OFF_VWO, lane, 0, nt, 96);
        vo = __builtin_amdgcn_wmma_f32_16x16x32_f16(false, a, false, b,
                                                    (short)0, vo, false, false);
#pragma unroll
        for (int r = 0; r < 8; ++r) {
          const int m = ((lane >> 4) << 3) + r;
          cvv[nt][r] += facv[wave * 16 + m] * vo[r];
        }
      }
    }
    __syncthreads();

    // ---- e-net: (t, path[0..120], 1) -> 32 -> 32 -> 32 -> 32 -> scalar ----
    mlp_layer<4, 2, true>(eIn, 128, wseg + OFF_EW0, 32, bA, 64, lane);
    mlp_layer<1, 2, true>(bA, 64, wseg + OFF_EH + 0 * 1024, 32, bB, 64, lane);
    mlp_layer<1, 2, true>(bB, 64, wseg + OFF_EH + 1 * 1024, 32, bA, 64, lane);
    mlp_layer<1, 2, true>(bA, 64, wseg + OFF_EH + 2 * 1024, 32, bB, 64, lane);
    mlp_out_col0<1>(bB, 64, wseg + OFF_EWO, eouts + wave * 16, lane);
    __syncthreads();

    if (lane < 16) {  // scalar phase C: cv_e, path append, state advance
      cv_e += facv[sL] * eouts[sL];
      einp[sL * 128 + 1 + i] = (_Float16)S_new;
      S_old = S_new;
    }

    if ((i % PERIOD) == 0) {  // maturity: vanilla payoff mean/var reduction
      const int ind_m = i / PERIOD - 1;
      const float disc_i = expf(-RRATE * (i * hstep));
#pragma unroll
      for (int t = 0; t < 6; ++t) {
        const int c = t * 16 + (lane & 15);
        const int srel = c - NSTR * ind_m;
        if (srel >= 0 && srel < NSTR) {
          const float Kstr = 0.8f + 0.02f * srel;
          float s1 = 0.f, s2 = 0.f;
#pragma unroll
          for (int r = 0; r < 8; ++r) {
            const int m = ((lane >> 4) << 3) + r;
            const float Sn = snew[wave * 16 + m];
            const float pay = relu1(Sn - Kstr);
            const float pr = disc_i * pay - cvv[t][r];
            s1 += pr;
            s2 += pr * pr;
          }
          atomicAdd(&acc[ind_m * NSTR + srel], s1);
          atomicAdd(&acc[84 + ind_m * NSTR + srel], s2);
        }
      }
    }
    __syncthreads();
  }

  if (lane < 16) {  // exotic payoff + per-sample outputs
    const float discT = expf(-RRATE * 1.0f);
    const float eV = discT * (run_max - S_old);
    wsEval[gs] = eV;
    wsCve[gs] = cv_e;
    const float pe = eV - cv_e;
    atomicAdd(&acc[168], eV);
    atomicAdd(&acc[169], pe);
    atomicAdd(&acc[170], pe * pe);
  }
}

// ------------------------------ output kernels ------------------------------
__global__ void finalize_k(const float* __restrict__ acc, float* __restrict__ out) {
  const int t = threadIdx.x;
  const float invMC = 1.0f / MC;
  if (t < 84) {
    const float s1 = acc[t], s2 = acc[84 + t];
    out[t] = s1 * invMC;                                    // pv
    out[84 + t] = (s2 - s1 * s1 * invMC) / (MC - 1);        // pvv (ddof=1)
  } else if (t == 84) {
    const float s1 = acc[169], s2 = acc[170];
    out[16552] = s1 * invMC;                                // mean(price_exo)
    out[16553] = (s2 - s1 * s1 * invMC) / (MC - 1);         // var(price_exo)
  }
}

__global__ void per_sample_k(const float* __restrict__ acc,
                             const float* __restrict__ eVal,
                             const float* __restrict__ cvE,
                             float* __restrict__ out) {
  const int s = blockIdx.x * blockDim.x + threadIdx.x;
  if (s < MC) {
    const float meanE = acc[168] / MC;
    const float ev = eVal[s], ce = cvE[s];
    out[168 + s] = ev - ce;                 // price_exo
    out[16554 + s] = ev - meanE - ce;       // error
  }
}

// -------------------------------- launcher ---------------------------------
extern "C" void kernel_launch(void* const* d_in, const int* in_sizes, int n_in,
                              void* d_out, int out_size, void* d_ws, size_t ws_size,
                              hipStream_t stream) {
  (void)in_sizes; (void)n_in; (void)out_size; (void)ws_size;
  const float* S0  = (const float*)d_in[0];
  const float* z   = (const float*)d_in[1];
  const float* dW0 = (const float*)d_in[2];
  const float* db0 = (const float*)d_in[3];
  const float* dhW = (const float*)d_in[4];
  const float* dhb = (const float*)d_in[5];
  const float* dWo = (const float*)d_in[6];
  const float* dbo = (const float*)d_in[7];
  const float* vW0 = (const float*)d_in[8];
  const float* vb0 = (const float*)d_in[9];
  const float* vhW = (const float*)d_in[10];
  const float* vhb = (const float*)d_in[11];
  const float* vWo = (const float*)d_in[12];
  const float* vbo = (const float*)d_in[13];
  const float* eW0 = (const float*)d_in[14];
  const float* eb0 = (const float*)d_in[15];
  const float* ehW = (const float*)d_in[16];
  const float* ehb = (const float*)d_in[17];
  const float* eWo = (const float*)d_in[18];
  const float* ebo = (const float*)d_in[19];

  char* ws = (char*)d_ws;
  _Float16* wsW = (_Float16*)(ws + WS_W);
  float* acc  = (float*)(ws + WS_ACC);
  float* eVal = (float*)(ws + WS_EVAL);
  float* cvE  = (float*)(ws + WS_CVE);
  float* out  = (float*)d_out;

  init_acc<<<dim3(1), dim3(256), 0, stream>>>(acc);
  prep_weights<<<dim3(4), dim3(256), 0, stream>>>(
      dW0, db0, dhW, dhb, dWo, dbo, vW0, vb0, vhW, vhb, vWo, vbo,
      eW0, eb0, ehW, ehb, eWo, ebo, wsW);
  mc_forward<<<dim3(MC / 64), dim3(128), LDS_TOTAL, stream>>>(
      S0, z, wsW, acc, eVal, cvE);
  finalize_k<<<dim3(1), dim3(128), 0, stream>>>(acc, out);
  per_sample_k<<<dim3(MC / 256), dim3(256), 0, stream>>>(acc, eVal, cvE, out);
}
